// ConMGIN_27384711480023
// MI455X (gfx1250) — compile-verified, matrix operand order checked by reading, not run on hardware
//
#include <hip/hip_runtime.h>
#include <hip/hip_bf16.h>
#include <math.h>

// ---------------------------------------------------------------------------
// Types for WMMA
// ---------------------------------------------------------------------------
typedef __attribute__((ext_vector_type(16))) __bf16 v16bf;
typedef __attribute__((ext_vector_type(8)))  __bf16 bf16x8;
typedef __attribute__((ext_vector_type(2)))  __bf16 bf16x2;
typedef __attribute__((ext_vector_type(8)))  float  v8f;

// f32 -> bf16 pair, packed into one dword. Uses native fptrunc (RTE) so the
// backend can select the packed bf16 convert instead of scalar bit-twiddling.
static __device__ __forceinline__ unsigned pk2bf(float lo, float hi) {
    bf16x2 p;
    p.x = (__bf16)lo;
    p.y = (__bf16)hi;
    return __builtin_bit_cast(unsigned, p);
}

static __device__ __forceinline__ v16bf join8(bf16x8 lo, bf16x8 hi) {
    return __builtin_shufflevector(lo, hi, 0,1,2,3,4,5,6,7,8,9,10,11,12,13,14,15);
}

// ---------------------------------------------------------------------------
// zero fill
// ---------------------------------------------------------------------------
__global__ void zero_kernel(float* __restrict__ p, long long n4) {
    long long i = (long long)blockIdx.x * blockDim.x + threadIdx.x;
    long long stride = (long long)gridDim.x * blockDim.x;
    for (; i < n4; i += stride) {
        ((float4*)p)[i] = make_float4(0.f, 0.f, 0.f, 0.f);
    }
}

// ---------------------------------------------------------------------------
// scatter-add: agg[dst] += feat[src], feature width F (multiple of 4).
// float4 gather + 4 relaxed agent-scope f32 atomics -> global_atomic_add_f32.
// Gathered matrices are L2-resident (102 MB / 51 MB vs 192 MB L2).
// ---------------------------------------------------------------------------
__global__ void scatter_add_kernel(const float* __restrict__ feat,
                                   const int* __restrict__ src,
                                   const int* __restrict__ dst,
                                   float* __restrict__ agg,
                                   long long total4, int F, int fshift, int fmask) {
    long long i = (long long)blockIdx.x * blockDim.x + threadIdx.x;
    long long stride = (long long)gridDim.x * blockDim.x;
    for (; i < total4; i += stride) {
        long long e = i >> fshift;
        int c = (int)(i & fmask) << 2;
        int s = src[e], d = dst[e];
        float4 v = *(const float4*)(feat + (size_t)s * F + c);
        float* q = agg + (size_t)d * F + c;
        __hip_atomic_fetch_add(q + 0, v.x, __ATOMIC_RELAXED, __HIP_MEMORY_SCOPE_AGENT);
        __hip_atomic_fetch_add(q + 1, v.y, __ATOMIC_RELAXED, __HIP_MEMORY_SCOPE_AGENT);
        __hip_atomic_fetch_add(q + 2, v.z, __ATOMIC_RELAXED, __HIP_MEMORY_SCOPE_AGENT);
        __hip_atomic_fetch_add(q + 3, v.w, __ATOMIC_RELAXED, __HIP_MEMORY_SCOPE_AGENT);
    }
}

// ---------------------------------------------------------------------------
// WMMA bf16 GEMM:  C[M,Nc] = epilogue( (A (+A2)) @ W[K,Nc] + bias )
//   block tile 128x64, 256 threads = 8 waves (4x2), wave tile 32x32 (2x2 frags)
//   K tiled by 32 (v_wmma_f32_16x16x32_bf16), double-buffered LDS,
//   software-pipelined global loads, one barrier per k-tile.
// Epilogues: 0 bias, 1 bias+relu, 2 bias+BN+relu, 3 sigmoid, 4 softplus-clip,
//            5 exp-clip
// ---------------------------------------------------------------------------
#define BM 128
#define BN 64
#define KT 32
#define LDA 40
#define LDB 40

template <int EPI, int HASA2>
__global__ __launch_bounds__(256)
void gemm_wmma_kernel(const float* __restrict__ A, const float* __restrict__ A2,
                      const float* __restrict__ W, const float* __restrict__ bias,
                      const float* __restrict__ e0, const float* __restrict__ e1p,
                      const float* __restrict__ e2, const float* __restrict__ e3,
                      float* __restrict__ C, int M, int K, int Nc) {
    __shared__ alignas(16) __bf16 sA[2][BM][LDA];
    __shared__ alignas(16) __bf16 sB[2][BN][LDB];

    const int tid  = threadIdx.x;
    const int lane = tid & 31;
    const int wave = tid >> 5;     // 0..7
    const int wr   = wave >> 1;    // 0..3
    const int wc   = wave & 1;     // 0..1
    const int rowBase = blockIdx.x * BM;
    const int colBase = blockIdx.y * BN;

    // A staging: 4 tasks/thread, task = (row, 4-wide k group)
    const int ar  = tid >> 3;           // base row (stepped by +32 per task)
    const int acg = (tid & 7) << 2;     // k-group offset
    // B staging: 1 task/thread = (2 adjacent k-rows, 4 adjacent cols)
    const int bkr = (tid >> 4) << 1;    // k-row pair base: 0,2,..,30
    const int bcg = (tid & 15) << 2;    // col group

    v8f acc[2][2];
#pragma unroll
    for (int mi = 0; mi < 2; ++mi)
#pragma unroll
        for (int ni = 0; ni < 2; ++ni) acc[mi][ni] = {};

    const int lr = lane & 15;
    const int kh = (lane >> 4) * 8;     // which 8-wide K chunk this lane holds

    float4 ra[4];
    float4 rb0, rb1;

    // ---- tile loader (global -> regs) ----
    auto load_tile = [&](int k0) {
#pragma unroll
        for (int i = 0; i < 4; ++i) {
            int grow = rowBase + ar + i * 32;
            float4 v = make_float4(0.f, 0.f, 0.f, 0.f);
            if (grow < M) {
                v = *(const float4*)(A + (size_t)grow * K + k0 + acg);
                if (HASA2) {
                    float4 w2 = *(const float4*)(A2 + (size_t)grow * K + k0 + acg);
                    v.x += w2.x; v.y += w2.y; v.z += w2.z; v.w += w2.w;
                }
            }
            ra[i] = v;
        }
        rb0 = *(const float4*)(W + (size_t)(k0 + bkr)     * Nc + colBase + bcg);
        rb1 = *(const float4*)(W + (size_t)(k0 + bkr + 1) * Nc + colBase + bcg);
    };
    // ---- commit (regs -> bf16 LDS, packed dword stores) ----
    auto commit_tile = [&](int buf) {
#pragma unroll
        for (int i = 0; i < 4; ++i) {
            *reinterpret_cast<uint2*>(&sA[buf][ar + i * 32][acg]) =
                make_uint2(pk2bf(ra[i].x, ra[i].y), pk2bf(ra[i].z, ra[i].w));
        }
        *reinterpret_cast<unsigned*>(&sB[buf][bcg + 0][bkr]) = pk2bf(rb0.x, rb1.x);
        *reinterpret_cast<unsigned*>(&sB[buf][bcg + 1][bkr]) = pk2bf(rb0.y, rb1.y);
        *reinterpret_cast<unsigned*>(&sB[buf][bcg + 2][bkr]) = pk2bf(rb0.z, rb1.z);
        *reinterpret_cast<unsigned*>(&sB[buf][bcg + 3][bkr]) = pk2bf(rb0.w, rb1.w);
    };

    const int nk = K / KT;
    load_tile(0);
    commit_tile(0);
    __syncthreads();

    for (int t = 0; t < nk; ++t) {
        const int cur = t & 1;
        if (t + 1 < nk) load_tile((t + 1) * KT);   // vmem in flight over the WMMAs

        // ---- fragments + WMMA (A lane: row lr, K chunks kh and 16+kh) ----
        v16bf af[2], bfr_[2];
#pragma unroll
        for (int mi = 0; mi < 2; ++mi) {
            int r = wr * 32 + mi * 16 + lr;
            bf16x8 lo = *(const bf16x8*)&sA[cur][r][kh];
            bf16x8 hi = *(const bf16x8*)&sA[cur][r][16 + kh];
            af[mi] = join8(lo, hi);
        }
#pragma unroll
        for (int ni = 0; ni < 2; ++ni) {
            int cc = wc * 32 + ni * 16 + lr;
            bf16x8 lo = *(const bf16x8*)&sB[cur][cc][kh];
            bf16x8 hi = *(const bf16x8*)&sB[cur][cc][16 + kh];
            bfr_[ni] = join8(lo, hi);
        }
#pragma unroll
        for (int mi = 0; mi < 2; ++mi)
#pragma unroll
            for (int ni = 0; ni < 2; ++ni)
                acc[mi][ni] = __builtin_amdgcn_wmma_f32_16x16x32_bf16(
                    false, af[mi], false, bfr_[ni], (short)0, acc[mi][ni], false, false);

        if (t + 1 < nk) {
            commit_tile((t + 1) & 1);   // writes the buffer not being read
            __syncthreads();            // single barrier per k-tile
        }
    }

    // ---- epilogue + store. C/D layout: row = base + 8*(lane>=16) + vgpr, col = lane&15
    const int vh = (lane >> 4) * 8;
#pragma unroll
    for (int mi = 0; mi < 2; ++mi) {
#pragma unroll
        for (int ni = 0; ni < 2; ++ni) {
            int col = colBase + wc * 32 + ni * 16 + lr;
#pragma unroll
            for (int v = 0; v < 8; ++v) {
                int row = rowBase + wr * 32 + mi * 16 + vh + v;
                if (row < M) {
                    float t = acc[mi][ni][v] + bias[col];
                    if (EPI == 1) {
                        t = fmaxf(t, 0.f);
                    } else if (EPI == 2) {
                        t = (t - e0[col]) * rsqrtf(e1p[col] + 1e-5f) * e2[col] + e3[col];
                        t = fmaxf(t, 0.f);
                    } else if (EPI == 3) {
                        t = 1.f / (1.f + __expf(-t));
                    } else if (EPI == 4) {
                        float sp = (t > 20.f) ? t : log1pf(__expf(t));
                        t = fminf(fmaxf(sp, 1e-4f), 1e4f);
                    } else if (EPI == 5) {
                        t = fminf(fmaxf(__expf(t), 1e-5f), 1e6f);
                    }
                    C[(size_t)row * Nc + col] = t;
                }
            }
        }
    }
}

// ---------------------------------------------------------------------------
// SpatialAttention + combine: per node n with z0=e1, z1=(e1+e2)/2, z2=e2:
//   w_k = tanh(z_k @ Wa1 + ba1) @ Wa2 ; beta = softmax(w) ; out = sum beta_k z_k
// ---------------------------------------------------------------------------
__global__ __launch_bounds__(256)
void attn_combine_kernel(const float* __restrict__ e1, const float* __restrict__ e2,
                         const float* __restrict__ Wa1, const float* __restrict__ ba1,
                         const float* __restrict__ Wa2, float* __restrict__ out, int n) {
    __shared__ float sW[64 * 16];
    __shared__ float sb[16];
    __shared__ float sw2[16];
    for (int i = threadIdx.x; i < 1024; i += 256) sW[i] = Wa1[i];
    if (threadIdx.x < 16) { sb[threadIdx.x] = ba1[threadIdx.x]; sw2[threadIdx.x] = Wa2[threadIdx.x]; }
    __syncthreads();

    int nid = blockIdx.x * 256 + threadIdx.x;
    if (nid >= n) return;
    const float* p1 = e1 + (size_t)nid * 64;
    const float* p2 = e2 + (size_t)nid * 64;

    float s0[16], s1[16], s2[16];
#pragma unroll
    for (int j = 0; j < 16; ++j) { s0[j] = sb[j]; s1[j] = sb[j]; s2[j] = sb[j]; }
    for (int i = 0; i < 64; ++i) {
        float a = p1[i], b = p2[i], c = 0.5f * (a + b);
        const float* wr = &sW[i * 16];
#pragma unroll
        for (int j = 0; j < 16; ++j) {
            float w = wr[j];
            s0[j] += a * w; s1[j] += c * w; s2[j] += b * w;
        }
    }
    float w0 = 0.f, w1 = 0.f, w2 = 0.f;
#pragma unroll
    for (int j = 0; j < 16; ++j) {
        w0 += tanhf(s0[j]) * sw2[j];
        w1 += tanhf(s1[j]) * sw2[j];
        w2 += tanhf(s2[j]) * sw2[j];
    }
    float m  = fmaxf(w0, fmaxf(w1, w2));
    float x0 = __expf(w0 - m), x1 = __expf(w1 - m), x2 = __expf(w2 - m);
    float inv = 1.f / (x0 + x1 + x2);
    float b0 = x0 * inv, b1 = x1 * inv, b2 = x2 * inv;

    float* po = out + (size_t)nid * 64;
    for (int i = 0; i < 64; ++i) {
        float a = p1[i], b = p2[i], c = 0.5f * (a + b);
        po[i] = b0 * a + b1 * c + b2 * b;
    }
}

// ---------------------------------------------------------------------------
// host side
// ---------------------------------------------------------------------------
static inline void run_gemm(int epi, const float* A, const float* A2, const float* W,
                            const float* bias, const float* e0, const float* e1p,
                            const float* e2, const float* e3, float* C,
                            int M, int K, int Nc, hipStream_t stream) {
    dim3 grid((M + BM - 1) / BM, Nc / BN);
    dim3 block(256);
    switch (epi) {
        case 0: gemm_wmma_kernel<0,0><<<grid, block, 0, stream>>>(A, A2, W, bias, e0, e1p, e2, e3, C, M, K, Nc); break;
        case 1:
            if (A2) gemm_wmma_kernel<1,1><<<grid, block, 0, stream>>>(A, A2, W, bias, e0, e1p, e2, e3, C, M, K, Nc);
            else    gemm_wmma_kernel<1,0><<<grid, block, 0, stream>>>(A, A2, W, bias, e0, e1p, e2, e3, C, M, K, Nc);
            break;
        case 2: gemm_wmma_kernel<2,0><<<grid, block, 0, stream>>>(A, A2, W, bias, e0, e1p, e2, e3, C, M, K, Nc); break;
        case 3: gemm_wmma_kernel<3,0><<<grid, block, 0, stream>>>(A, A2, W, bias, e0, e1p, e2, e3, C, M, K, Nc); break;
        case 4: gemm_wmma_kernel<4,0><<<grid, block, 0, stream>>>(A, A2, W, bias, e0, e1p, e2, e3, C, M, K, Nc); break;
        default: gemm_wmma_kernel<5,0><<<grid, block, 0, stream>>>(A, A2, W, bias, e0, e1p, e2, e3, C, M, K, Nc); break;
    }
}

extern "C" void kernel_launch(void* const* d_in, const int* in_sizes, int n_in,
                              void* d_out, int out_size, void* d_ws, size_t ws_size,
                              hipStream_t stream) {
    const float* x    = (const float*)d_in[0];
    const int*   sadj = (const int*)d_in[1];
    const int*   fadj = (const int*)d_in[2];
    const float* Wg1a = (const float*)d_in[3];  const float* bg1a = (const float*)d_in[4];
    const float* Wg1b = (const float*)d_in[5];  const float* bg1b = (const float*)d_in[6];
    const float* Wg2a = (const float*)d_in[7];  const float* bg2a = (const float*)d_in[8];
    const float* Wg2b = (const float*)d_in[9];  const float* bg2b = (const float*)d_in[10];
    const float* Wa1  = (const float*)d_in[11]; const float* ba1  = (const float*)d_in[12];
    const float* Wa2  = (const float*)d_in[13];
    const float* Wmlp = (const float*)d_in[14]; const float* bmlp = (const float*)d_in[15];
    const float* Wd   = (const float*)d_in[16]; const float* bd   = (const float*)d_in[17];
    const float* bn_g = (const float*)d_in[18]; const float* bn_b = (const float*)d_in[19];
    const float* bn_m = (const float*)d_in[20]; const float* bn_v = (const float*)d_in[21];
    const float* Wpi  = (const float*)d_in[22]; const float* bpi  = (const float*)d_in[23];
    const float* Wv   = (const float*)d_in[24]; const float* bv   = (const float*)d_in[25];
    const float* Wmu  = (const float*)d_in[26]; const float* bmu  = (const float*)d_in[27];

    const int NF = 512, H1 = 256, H2 = 64;
    const int n  = in_sizes[0] / NF;      // 50000
    const int Eg = in_sizes[1] / 2;       // 1.6M

    // workspace layout (floats): agg[n*512] | t1[n*256] | h1[n*256]
    size_t need = ((size_t)n * 1024) * sizeof(float);
    if (ws_size < need) return;
    float* ws  = (float*)d_ws;
    float* agg = ws;
    float* t1  = ws + (size_t)n * 512;
    float* h1  = t1 + (size_t)n * 256;
    float* t2      = t1;   // reuse (n*64) after t1 consumed
    float* embcomb = t1;   // reuse in post-GIN phase
    float* hbuf    = h1;   // reuse in post-GIN phase

    float* out  = (float*)d_out;
    float* emb1 = out;
    float* emb2 = out + (size_t)n * 64;
    float* embO = out + (size_t)n * 128;
    float* piO  = out + (size_t)n * 192;
    float* varO = piO  + (size_t)n * 512;
    float* muO  = varO + (size_t)n * 512;

    const int* edges[2]   = { sadj, fadj };
    float*     embOut[2]  = { emb1, emb2 };

    for (int g = 0; g < 2; ++g) {
        const int* src = edges[g];
        const int* dst = edges[g] + Eg;

        // ---- GIN conv1 (feature width 512) ----
        long long n4 = (long long)n * 512 / 4;
        zero_kernel<<<dim3((unsigned)((n4 + 255) / 256)), dim3(256), 0, stream>>>(agg, n4);
        long long tot4 = (long long)Eg * (512 / 4);
        unsigned sblocks = (unsigned)(((tot4 + 255) / 256 < (1 << 20)) ? (tot4 + 255) / 256 : (1 << 20));
        scatter_add_kernel<<<dim3(sblocks), dim3(256), 0, stream>>>(x, src, dst, agg, tot4, 512, 7, 127);
        // t1 = relu((x+agg) @ Wg1a + bg1a)
        run_gemm(1, x, agg, Wg1a, bg1a, nullptr, nullptr, nullptr, nullptr, t1, n, NF, H1, stream);
        // h1 = relu(t1 @ Wg1b + bg1b)   (outer relu of _gin fused)
        run_gemm(1, t1, nullptr, Wg1b, bg1b, nullptr, nullptr, nullptr, nullptr, h1, n, H1, H1, stream);

        // ---- GIN conv2 (feature width 256) ----
        n4 = (long long)n * 256 / 4;
        zero_kernel<<<dim3((unsigned)((n4 + 255) / 256)), dim3(256), 0, stream>>>(agg, n4);
        tot4 = (long long)Eg * (256 / 4);
        sblocks = (unsigned)(((tot4 + 255) / 256 < (1 << 20)) ? (tot4 + 255) / 256 : (1 << 20));
        scatter_add_kernel<<<dim3(sblocks), dim3(256), 0, stream>>>(h1, src, dst, agg, tot4, 256, 6, 63);
        // t2 = relu((h1+agg) @ Wg2a + bg2a)
        run_gemm(1, h1, agg, Wg2a, bg2a, nullptr, nullptr, nullptr, nullptr, t2, n, H1, H2, stream);
        // emb_g = t2 @ Wg2b + bg2b      (no activation)
        run_gemm(0, t2, nullptr, Wg2b, bg2b, nullptr, nullptr, nullptr, nullptr, embOut[g], n, H2, H2, stream);
    }

    // ---- attention combine -> embcomb [n,64] ----
    attn_combine_kernel<<<dim3((n + 255) / 256), dim3(256), 0, stream>>>(
        emb1, emb2, Wa1, ba1, Wa2, embcomb, n);

    // ---- emb = embcomb @ Wmlp + bmlp ----
    run_gemm(0, embcomb, nullptr, Wmlp, bmlp, nullptr, nullptr, nullptr, nullptr, embO, n, H2, H2, stream);
    // ---- h = relu(BN(emb @ Wd + bd)) ----
    run_gemm(2, embO, nullptr, Wd, bd, bn_m, bn_v, bn_g, bn_b, hbuf, n, H2, H1, stream);
    // ---- heads ----
    run_gemm(3, hbuf, nullptr, Wpi, bpi, nullptr, nullptr, nullptr, nullptr, piO,  n, H1, NF, stream);
    run_gemm(4, hbuf, nullptr, Wv,  bv,  nullptr, nullptr, nullptr, nullptr, varO, n, H1, NF, stream);
    run_gemm(5, hbuf, nullptr, Wmu, bmu, nullptr, nullptr, nullptr, nullptr, muO,  n, H1, NF, stream);
}
